// ContextFeatureExtractor_45749991637040
// MI455X (gfx1250) — matrix operand order repaired and checked by reading, not compiled, not run on hardware
//
#include <hip/hip_runtime.h>
#include <hip/hip_bf16.h>

typedef _Float16 half8_t __attribute__((ext_vector_type(8)));
typedef _Float16 v16h    __attribute__((ext_vector_type(16)));
typedef float    v8f     __attribute__((ext_vector_type(8)));

union AFrag { v16h v; half8_t h[2]; };

#define NB   64
#define HH   256
#define WW   256

// gfx1250 async global->LDS copy (ASYNCcnt-tracked, bypasses VGPRs).
// lds generic pointer: low 32 bits are the LDS byte offset (LDS aperture rule).
__device__ __forceinline__ void async_ld_b128(void* lds_ptr, const void* gptr) {
    unsigned loff = (unsigned)(unsigned long long)lds_ptr;
    asm volatile("global_load_async_to_lds_b128 %0, %1, off"
                 :: "v"(loff), "v"(gptr) : "memory");
}
__device__ __forceinline__ void wait_async0() {
    asm volatile("s_wait_asynccnt 0" ::: "memory");
}

// ---------------------------------------------------------------------------
// Kernel 1: weight packing
//   lut16   [10][9][32] f16 : lut[c][t][o] = conv1_w[o][c][t]
//   wpack2  [9][64][32] f16 : w[t][o][c]   = conv2_w[o][c][t]
//   gfc16   [128][4096] f16 : gfc_w cast
// ---------------------------------------------------------------------------
__global__ __launch_bounds__(256) void pack_weights_k(
    const float* __restrict__ conv1_w, const float* __restrict__ conv2_w,
    const float* __restrict__ gfc_w,
    _Float16* __restrict__ lut16, _Float16* __restrict__ wpack2,
    _Float16* __restrict__ gfc16)
{
    int idx = blockIdx.x * blockDim.x + threadIdx.x;
    int stride = gridDim.x * blockDim.x;
    for (int i = idx; i < 2880; i += stride) {
        int o = i & 31; int t = (i >> 5) % 9; int c = i / 288;
        lut16[i] = (_Float16)conv1_w[(o * 10 + c) * 9 + t];
    }
    for (int i = idx; i < 18432; i += stride) {
        int c = i & 31; int o = (i >> 5) & 63; int t = i >> 11;
        wpack2[i] = (_Float16)conv2_w[(o * 32 + c) * 9 + t];
    }
    for (int i = idx; i < 128 * 4096; i += stride) {
        gfc16[i] = (_Float16)gfc_w[i];
    }
}

// ---------------------------------------------------------------------------
// Kernel 2: slot statistics (per-color count + coord means) -> stats[b][40]
// ---------------------------------------------------------------------------
__global__ __launch_bounds__(256) void slot_stats_k(
    const int* __restrict__ grid, float* __restrict__ stats)
{
    __shared__ float s_cnt[10], s_sy[10], s_sx[10];
    const int b = blockIdx.x, tid = threadIdx.x;
    if (tid < 10) { s_cnt[tid] = 0.f; s_sy[tid] = 0.f; s_sx[tid] = 0.f; }
    __syncthreads();
    float c[10], sy[10], sx[10];
#pragma unroll
    for (int k = 0; k < 10; ++k) { c[k] = 0.f; sy[k] = 0.f; sx[k] = 0.f; }
    for (int p = tid; p < HH * WW; p += 256) {
        int y = p >> 8, x = p & 255;
        int col = grid[b * HH * WW + p];
        float fy = (float)y, fx = (float)x;
#pragma unroll
        for (int k = 0; k < 10; ++k) {
            float m = (col == k) ? 1.f : 0.f;
            c[k] += m; sy[k] += m * fy; sx[k] += m * fx;
        }
    }
#pragma unroll
    for (int k = 0; k < 10; ++k) {
        atomicAdd(&s_cnt[k], c[k]);
        atomicAdd(&s_sy[k], sy[k]);
        atomicAdd(&s_sx[k], sx[k]);
    }
    __syncthreads();
    if (tid < 10) {
        float cnt = s_cnt[tid];
        float my = cnt > 0.f ? s_sy[tid] / cnt : 0.f;
        float mx = cnt > 0.f ? s_sx[tid] / cnt : 0.f;
        float* o = stats + b * 40 + tid * 4;
        o[0] = cnt; o[1] = cnt; o[2] = my; o[3] = mx;
    }
}

// ---------------------------------------------------------------------------
// Kernel 3: fused conv1(f16 LUT) + conv2(WMMA f16) + ReLU + 32x32 avg-pool
//   One workgroup per (batch, pooled 8x8 block). 256 threads = 8 wave32.
// ---------------------------------------------------------------------------
__global__ __launch_bounds__(256) void fused_conv_pool_k(
    const int* __restrict__ grid,
    const _Float16* __restrict__ lut16,      // [10*9*32] f16
    const float* __restrict__ conv1_b,       // [32]
    const _Float16* __restrict__ wpack2,     // [9*64*32] f16
    const float* __restrict__ conv2_b,       // [64]
    _Float16* __restrict__ pooled16)         // [64][4096]  k = c*64 + py*8 + px
{
    __shared__ __align__(16) _Float16 s_lut[2880];
    __shared__ __align__(16) _Float16 s_b1h[32];
    __shared__ int s_col[36 * 36];
    __shared__ __align__(16) _Float16 s_h1[34 * 34 * 32];
    __shared__ __align__(16) _Float16 s_w[9 * 64 * 32];
    __shared__ float s_pool[64];

    const int tid = threadIdx.x;
    const int b = blockIdx.x >> 6;
    const int blk = blockIdx.x & 63;
    const int y0 = (blk >> 3) * 32, x0 = (blk & 7) * 32;

    // ---- stage 0: async-copy conv2 weights (36KB) and f16 LUT (5.76KB) to LDS
    for (int i = tid; i < 2304; i += 256)          // 36864 B / 16
        async_ld_b128((char*)s_w + i * 16, (const char*)wpack2 + i * 16);
    for (int i = tid; i < 360; i += 256)           // 5760 B / 16
        async_ld_b128((char*)s_lut + i * 16, (const char*)lut16 + i * 16);
    if (tid < 32) s_b1h[tid] = (_Float16)conv1_b[tid];
    if (tid < 64) s_pool[tid] = 0.f;
    // stage 36x36 color tile (OOB = -1)
    for (int i = tid; i < 36 * 36; i += 256) {
        int cy = i / 36, cx = i % 36;
        int gy = y0 + cy - 2, gx = x0 + cx - 2;
        s_col[i] = ((unsigned)gy < (unsigned)HH && (unsigned)gx < (unsigned)WW)
                       ? grid[(b * HH + gy) * WW + gx] : -1;
    }
    wait_async0();
    __syncthreads();

    // ---- stage 1: conv1 via f16 LUT gather into 34x34x32 f16 tile (zero-pad)
    for (int p = tid; p < 34 * 34; p += 256) {
        int ty = p / 34, tx = p % 34;
        int gy = y0 + ty - 1, gx = x0 + tx - 1;
        bool inimg = ((unsigned)gy < (unsigned)HH) && ((unsigned)gx < (unsigned)WW);
        _Float16* dst = &s_h1[p * 32];
        if (inimg) {
            half8_t acc8[4];
#pragma unroll
            for (int q = 0; q < 4; ++q)
                acc8[q] = *(const half8_t*)&s_b1h[q * 8];
#pragma unroll
            for (int t = 0; t < 9; ++t) {
                int col = s_col[(ty + t / 3) * 36 + (tx + t % 3)];
                if (col >= 0) {
                    const half8_t* lp = (const half8_t*)&s_lut[(col * 9 + t) * 32];
                    acc8[0] += lp[0]; acc8[1] += lp[1];
                    acc8[2] += lp[2]; acc8[3] += lp[3];
                }
            }
#pragma unroll
            for (int q = 0; q < 4; ++q) {
                half8_t v = acc8[q];
#pragma unroll
                for (int e = 0; e < 8; ++e)
                    v[e] = v[e] > (_Float16)0.f ? v[e] : (_Float16)0.f;
                *(half8_t*)(dst + q * 8) = v;
            }
        } else {
            half8_t z = {};
#pragma unroll
            for (int q = 0; q < 4; ++q)
                *(half8_t*)(dst + q * 8) = z;
        }
    }
    __syncthreads();

    // ---- stage 2: conv2 as implicit GEMM with v_wmma_f32_16x16x32_f16
    const int wave = tid >> 5;
    const int lane = tid & 31;
    const int lh = lane & 15;
    const int hi = lane >> 4;
    const int c0a = hi ? 8 : 0;    // A-frag K base (ISA 16-bit A 16x32 layout)
    const int c0b = hi ? 8 : 0;    // B-frag K base (same K order as A-frag)

    float bias_n[4];
#pragma unroll
    for (int n = 0; n < 4; ++n) bias_n[n] = conv2_b[n * 16 + lh];

    for (int s = wave; s < 64; s += 8) {
        const int py = s >> 1;
        const int px0 = (s & 1) * 16;
        v8f accv[4] = {};
#pragma unroll
        for (int t = 0; t < 9; ++t) {
            const int dy = t / 3, dx = t % 3;
            const _Float16* ap = &s_h1[((py + dy) * 34 + (px0 + lh + dx)) * 32];
            AFrag a;
            a.h[0] = *(const half8_t*)(ap + c0a);
            a.h[1] = *(const half8_t*)(ap + c0a + 16);
#pragma unroll
            for (int n = 0; n < 4; ++n) {
                const _Float16* bp = &s_w[(t * 64 + n * 16 + lh) * 32 + c0b];
                AFrag bb;
                bb.h[0] = *(const half8_t*)(bp);
                bb.h[1] = *(const half8_t*)(bp + 16);
                accv[n] = __builtin_amdgcn_wmma_f32_16x16x32_f16(
                    false, a.v, false, bb.v, (short)0, accv[n], false, false);
            }
        }
        // bias + relu + partial pool (sum over this strip's 8 rows per half)
#pragma unroll
        for (int n = 0; n < 4; ++n) {
            float ssum = 0.f;
#pragma unroll
            for (int r = 0; r < 8; ++r)
                ssum += fmaxf(accv[n][r] + bias_n[n], 0.f);
            atomicAdd(&s_pool[n * 16 + lh], ssum);
        }
    }
    __syncthreads();

    if (tid < 64)
        pooled16[b * 4096 + tid * 64 + blk] =
            (_Float16)(s_pool[tid] * (1.f / 1024.f));
}

// ---------------------------------------------------------------------------
// Kernel 4: grid_feat GEMM  [64 x 4096] x [4096 x 128]^T  via WMMA f16
// ---------------------------------------------------------------------------
__global__ __launch_bounds__(32) void gfc_gemm_k(
    const _Float16* __restrict__ pooled16, const _Float16* __restrict__ gfc16,
    const float* __restrict__ gfc_b, float* __restrict__ out)
{
    const int lane = threadIdx.x & 31;
    const int lh = lane & 15;
    const int hi = lane >> 4;
    const int m0 = (blockIdx.x >> 3) * 16;
    const int n0 = (blockIdx.x & 7) * 16;
    const int c0a = hi ? 8 : 0;
    const int c0b = hi ? 8 : 0;

    const _Float16* arow = pooled16 + (m0 + lh) * 4096;
    const _Float16* brow = gfc16 + (n0 + lh) * 4096;

    v8f acc = {};
    for (int k0 = 0; k0 < 4096; k0 += 32) {
        AFrag a, bb;
        a.h[0] = *(const half8_t*)(arow + k0 + c0a);
        a.h[1] = *(const half8_t*)(arow + k0 + c0a + 16);
        bb.h[0] = *(const half8_t*)(brow + k0 + c0b);
        bb.h[1] = *(const half8_t*)(brow + k0 + c0b + 16);
        acc = __builtin_amdgcn_wmma_f32_16x16x32_f16(
            false, a.v, false, bb.v, (short)0, acc, false, false);
    }
    float bias = gfc_b[n0 + lh];
#pragma unroll
    for (int r = 0; r < 8; ++r) {
        int row = m0 + hi * 8 + r;
        out[row * 304 + n0 + lh] = acc[r] + bias;
    }
}

// ---------------------------------------------------------------------------
// Kernel 5: tiny MLP heads (slot/rel/size/theme/prog); one block per row.
// ---------------------------------------------------------------------------
__global__ __launch_bounds__(64) void mlp_heads_k(
    const float* __restrict__ rel_in, const float* __restrict__ size_in,
    const float* __restrict__ theme_in, const float* __restrict__ stats,
    const float* __restrict__ sl1_w, const float* __restrict__ sl1_b,
    const float* __restrict__ sl2_w, const float* __restrict__ sl2_b,
    const float* __restrict__ sl3_w, const float* __restrict__ sl3_b,
    const float* __restrict__ rel1_w, const float* __restrict__ rel1_b,
    const float* __restrict__ rel2_w, const float* __restrict__ rel2_b,
    const float* __restrict__ sz1_w, const float* __restrict__ sz1_b,
    const float* __restrict__ sz2_w, const float* __restrict__ sz2_b,
    const float* __restrict__ th1_w, const float* __restrict__ th1_b,
    const float* __restrict__ th2_w, const float* __restrict__ th2_b,
    const float* __restrict__ pr1_b,
    const float* __restrict__ pr2_w, const float* __restrict__ pr2_b,
    const float* __restrict__ pr3_w, const float* __restrict__ pr3_b,
    float* __restrict__ out)
{
    __shared__ float xin[64], hA[64], hB[64];
    const int b = blockIdx.x, t = threadIdx.x;
    float* orow = out + b * 304;

    // ---- slot: 40 -> 64 -> 32 -> 32  (out cols 128..160)
    if (t < 40) xin[t] = stats[b * 40 + t];
    __syncthreads();
    {
        float s = sl1_b[t];
        for (int i = 0; i < 40; ++i) s += sl1_w[t * 40 + i] * xin[i];
        hA[t] = fmaxf(s, 0.f);
    }
    __syncthreads();
    if (t < 32) {
        float s = sl2_b[t];
        for (int i = 0; i < 64; ++i) s += sl2_w[t * 64 + i] * hA[i];
        hB[t] = fmaxf(s, 0.f);
    }
    __syncthreads();
    if (t < 32) {
        float s = sl3_b[t];
        for (int i = 0; i < 32; ++i) s += sl3_w[t * 32 + i] * hB[i];
        orow[128 + t] = s;
    }
    __syncthreads();

    // ---- rel: 64 -> 64 -> 64  (out cols 192..256)
    xin[t] = rel_in[b * 64 + t];
    __syncthreads();
    {
        float s = rel1_b[t];
        for (int i = 0; i < 64; ++i) s += rel1_w[t * 64 + i] * xin[i];
        hA[t] = fmaxf(s, 0.f);
    }
    __syncthreads();
    {
        float s = rel2_b[t];
        for (int i = 0; i < 64; ++i) s += rel2_w[t * 64 + i] * hA[i];
        orow[192 + t] = s;
    }
    __syncthreads();

    // ---- size: 4 -> 16 -> 16  (out cols 256..272)
    if (t < 4) xin[t] = size_in[b * 4 + t];
    __syncthreads();
    if (t < 16) {
        float s = sz1_b[t];
        for (int i = 0; i < 4; ++i) s += sz1_w[t * 4 + i] * xin[i];
        hA[t] = fmaxf(s, 0.f);
    }
    __syncthreads();
    if (t < 16) {
        float s = sz2_b[t];
        for (int i = 0; i < 16; ++i) s += sz2_w[t * 16 + i] * hA[i];
        orow[256 + t] = s;
    }
    __syncthreads();

    // ---- theme: 10 -> 32 -> 32  (out cols 272..304)
    if (t < 10) xin[t] = theme_in[b * 10 + t];
    __syncthreads();
    if (t < 32) {
        float s = th1_b[t];
        for (int i = 0; i < 10; ++i) s += th1_w[t * 10 + i] * xin[i];
        hA[t] = fmaxf(s, 0.f);
    }
    __syncthreads();
    if (t < 32) {
        float s = th2_b[t];
        for (int i = 0; i < 32; ++i) s += th2_w[t * 32 + i] * hA[i];
        orow[272 + t] = s;
    }
    __syncthreads();

    // ---- prog: input is all-zeros -> h1 = relu(pr1_b)  (out cols 160..192)
    hA[t] = fmaxf(pr1_b[t], 0.f);
    __syncthreads();
    if (t < 32) {
        float s = pr2_b[t];
        for (int i = 0; i < 64; ++i) s += pr2_w[t * 64 + i] * hA[i];
        hB[t] = fmaxf(s, 0.f);
    }
    __syncthreads();
    if (t < 32) {
        float s = pr3_b[t];
        for (int i = 0; i < 32; ++i) s += pr3_w[t * 32 + i] * hB[i];
        orow[160 + t] = s;
    }
}

// ---------------------------------------------------------------------------
extern "C" void kernel_launch(void* const* d_in, const int* in_sizes, int n_in,
                              void* d_out, int out_size, void* d_ws, size_t ws_size,
                              hipStream_t stream) {
    (void)in_sizes; (void)n_in; (void)out_size; (void)ws_size;
    // setup_inputs order: grid, rel_features, size_oracle, theme_priors, params(dict order)
    const int*   grid      = (const int*)  d_in[0];
    const float* rel_in    = (const float*)d_in[1];
    const float* size_in   = (const float*)d_in[2];
    const float* theme_in  = (const float*)d_in[3];
    const float* conv1_w = (const float*)d_in[4];
    const float* conv1_b = (const float*)d_in[5];
    const float* conv2_w = (const float*)d_in[6];
    const float* conv2_b = (const float*)d_in[7];
    const float* gfc_w   = (const float*)d_in[8];
    const float* gfc_b   = (const float*)d_in[9];
    const float* sl1_w = (const float*)d_in[10]; const float* sl1_b = (const float*)d_in[11];
    const float* sl2_w = (const float*)d_in[12]; const float* sl2_b = (const float*)d_in[13];
    const float* sl3_w = (const float*)d_in[14]; const float* sl3_b = (const float*)d_in[15];
    const float* rel1_w = (const float*)d_in[16]; const float* rel1_b = (const float*)d_in[17];
    const float* rel2_w = (const float*)d_in[18]; const float* rel2_b = (const float*)d_in[19];
    const float* sz1_w = (const float*)d_in[20]; const float* sz1_b = (const float*)d_in[21];
    const float* sz2_w = (const float*)d_in[22]; const float* sz2_b = (const float*)d_in[23];
    const float* th1_w = (const float*)d_in[24]; const float* th1_b = (const float*)d_in[25];
    const float* th2_w = (const float*)d_in[26]; const float* th2_b = (const float*)d_in[27];
    const float* pr1_b = (const float*)d_in[29];
    const float* pr2_w = (const float*)d_in[30]; const float* pr2_b = (const float*)d_in[31];
    const float* pr3_w = (const float*)d_in[32]; const float* pr3_b = (const float*)d_in[33];
    float* out = (float*)d_out;

    char* ws = (char*)d_ws;
    _Float16*  lut16    = (_Float16*) (ws + 0);        // 5760 B
    _Float16*  wpack2   = (_Float16*) (ws + 5888);     // 36864 B
    _Float16*  gfc16    = (_Float16*) (ws + 42752);    // 1048576 B
    _Float16*  pooled16 = (_Float16*) (ws + 1091328);  // 524288 B
    float*     stats    = (float*)    (ws + 1615616);  // 10240 B

    pack_weights_k<<<256, 256, 0, stream>>>(conv1_w, conv2_w, gfc_w,
                                            lut16, wpack2, gfc16);
    slot_stats_k<<<NB, 256, 0, stream>>>(grid, stats);
    fused_conv_pool_k<<<NB * 64, 256, 0, stream>>>(grid, lut16, conv1_b,
                                                   wpack2, conv2_b, pooled16);
    gfc_gemm_k<<<32, 32, 0, stream>>>(pooled16, gfc16, gfc_b, out);
    mlp_heads_k<<<NB, 64, 0, stream>>>(rel_in, size_in, theme_in, stats,
        sl1_w, sl1_b, sl2_w, sl2_b, sl3_w, sl3_b,
        rel1_w, rel1_b, rel2_w, rel2_b,
        sz1_w, sz1_b, sz2_w, sz2_b,
        th1_w, th1_b, th2_w, th2_b,
        pr1_b, pr2_w, pr2_b, pr3_w, pr3_b, out);
}
